// SoftmaxAttention_15315853377955
// MI455X (gfx1250) — matrix operand order repaired and checked by reading, not compiled, not run on hardware
//
#include <hip/hip_runtime.h>

// ---------------------------------------------------------------------------
// MI455X (gfx1250) causal MHA:  B=1, L=4096, D=1024, H=16, hd=64
// Compute-bound (~85 GFLOP vs ~50MB traffic) -> all GEMM-like ops via
// v_wmma_f32_16x16x32_bf16; GEMM A-tiles staged through the Tensor Data
// Mover, software-pipelined (issue-ahead + s_wait_tensorcnt 1) with a
// double-buffered LDS tile so the DMA overlaps the matrix math.
// ---------------------------------------------------------------------------

typedef __attribute__((ext_vector_type(16))) __bf16 v16bf;
typedef __attribute__((ext_vector_type(8)))  float  v8f;
typedef __attribute__((ext_vector_type(4)))  unsigned int v4u;
typedef __attribute__((ext_vector_type(8)))  int v8i;
typedef __attribute__((ext_vector_type(4)))  int v4i;

union AFrag { v16bf v; unsigned int u[8]; v4u q[2]; };

#define SEQ_L 4096
#define DIM_D 1024
#define N_H   16
#define HD    64

#if __has_builtin(__builtin_amdgcn_tensor_load_to_lds)
#define HAVE_TDM 1
#else
#define HAVE_TDM 0
#endif

__device__ __forceinline__ unsigned short f2bf(float f) {
  unsigned int u = __builtin_bit_cast(unsigned int, f);
  unsigned int r = 0x7FFFu + ((u >> 16) & 1u);   // round-to-nearest-even
  return (unsigned short)((u + r) >> 16);
}

#if HAVE_TDM
// Issue a TDM load of a 16-row x 32-col bf16 tile (row stride K elems) from
// X[row0][k0] into LDS at byte offset lds_off. Per-wave, tracked by TENSORcnt.
__device__ __forceinline__ void tdm_load_tile(
    const unsigned short* X, int row0, int k0, int K, int M,
    unsigned int lds_off) {
  unsigned long long ga = (unsigned long long)(uintptr_t)X +
                          ((((unsigned long long)row0 * (unsigned long long)K) +
                            (unsigned long long)k0) << 1);
  v4u g0;
  g0[0] = 1u;                                        // count=1, user D#
  g0[1] = lds_off;                                   // LDS byte address
  g0[2] = (unsigned int)ga;                          // global_addr[31:0]
  g0[3] = (unsigned int)((ga >> 32) & 0x01FFFFFFu)   // global_addr[56:32]
          | (2u << 30);                              // type=2 ("image")
  v8i g1;
  g1[0] = (int)(1u << 16);                           // data_size = 2 bytes
  g1[1] = (int)(((unsigned)K & 0xFFFFu) << 16);      // tensor_dim0[15:0]
  g1[2] = (int)((((unsigned)K >> 16) & 0xFFFFu)      // tensor_dim0[31:16]
          | (((unsigned)M & 0xFFFFu) << 16));        // tensor_dim1[15:0]
  g1[3] = (int)((((unsigned)M >> 16) & 0xFFFFu)      // tensor_dim1[31:16]
          | (32u << 16));                            // tile_dim0 = 32 elems
  g1[4] = 16;                                        // tile_dim1 = 16 rows
  g1[5] = (int)K;                                    // tensor_dim0_stride
  g1[6] = 0; g1[7] = 0;
  v4i gz = {0, 0, 0, 0};
#if __clang_major__ >= 23
  v8i gz8 = {0, 0, 0, 0, 0, 0, 0, 0};
  __builtin_amdgcn_tensor_load_to_lds(g0, g1, gz, gz, gz8, 0);
#else
  __builtin_amdgcn_tensor_load_to_lds(g0, g1, gz, gz, 0);
#endif
}
#endif

// --------------------------- fp32 -> bf16 conversion -----------------------
__global__ __launch_bounds__(256) void cvt_f32_to_bf16_k(
    const float* __restrict__ src, unsigned short* __restrict__ dst, int n) {
  int i = blockIdx.x * blockDim.x + threadIdx.x;
  if (i < n) dst[i] = f2bf(src[i]);
}

// --------------------------- bf16 WMMA GEMM:  Y = X * W^T ------------------
// X: [M x K] bf16 row-major, W: [N x K] bf16 row-major (torch Linear weight).
// Each wave: 16(M) x 64(N) output (4 accumulator tiles, A-frag reused 4x).
// Block = 4 waves -> 16 x 256 block tile. grid = (N/256, M/16).
// A-tile (16x32 bf16): TDM-loaded to LDS, double-buffered, issued one
// iteration ahead so the DMA overlaps the WMMAs (wait TENSORcnt<=1).
// mode 0: store bf16 as [H][L][64]   (Q, K)
// mode 1: store bf16 as [H][64][L]   (V transposed, for PV B-fragments)
// mode 2: store f32  as [M][N]       (final output)
__global__ __launch_bounds__(128) void gemm_bf16_k(
    const unsigned short* __restrict__ X, const unsigned short* __restrict__ W,
    int M, int K, int Nn, int mode,
    unsigned short* __restrict__ outb, float* __restrict__ outf) {
  extern __shared__ unsigned short smem[];   // 4 waves * 2 bufs * 512 halves = 8KB

  const int lane = threadIdx.x & 31;
  const int wave = threadIdx.x >> 5;
  const int hf   = lane >> 4;        // which 16-lane half
  const int l16  = lane & 15;
  const int tile_m = blockIdx.y * 16;
  const int tile_n = blockIdx.x * 256 + wave * 64;

  v8f acc[4];
  const v8f vzero = {0.f,0.f,0.f,0.f,0.f,0.f,0.f,0.f};
  acc[0] = vzero; acc[1] = vzero; acc[2] = vzero; acc[3] = vzero;

#if HAVE_TDM
  // prologue: kick off the first A tile
  tdm_load_tile(X, tile_m, 0, K, M, (unsigned int)(wave * 2048));
#endif

  for (int k0 = 0; k0 < K; k0 += 32) {
    AFrag a;
#if HAVE_TDM
    {
      const unsigned int buf = (unsigned int)((k0 >> 5) & 1);
      const unsigned int lds_off = (unsigned int)(wave * 2048) + buf * 1024u;
      if (k0 + 32 < K) {
        // issue next tile into the other buffer, then wait for the older one
        tdm_load_tile(X, tile_m, k0 + 32, K, M,
                      (unsigned int)(wave * 2048) + (buf ^ 1u) * 1024u);
        __builtin_amdgcn_s_wait_tensorcnt(1);
      } else {
        __builtin_amdgcn_s_wait_tensorcnt(0);
      }
      asm volatile("" ::: "memory");
      const unsigned short* aw = smem + (lds_off >> 1);
      a.q[0] = *(const v4u*)(aw + l16 * 32 + 8 * hf);        // K = 0..7  (+8*hf)
      a.q[1] = *(const v4u*)(aw + l16 * 32 + 16 + 8 * hf);   // K = 16..23(+8*hf)
    }
#else
#pragma unroll
    for (int i = 0; i < 8; ++i) {
      int kk = k0 + ((i < 4) ? 2 * i : 16 + 2 * (i - 4)) + 8 * hf;
      a.u[i] = *(const unsigned int*)(X + (size_t)(tile_m + l16) * K + kk);
    }
#endif
    // prefetch next B tile slice
    if (k0 + 32 < K)
      __builtin_prefetch(W + (size_t)(tile_n + l16) * K + k0 + 32 + 16 * hf, 0, 3);
#pragma unroll
    for (int t = 0; t < 4; ++t) {
      AFrag b;
#pragma unroll
      for (int i = 0; i < 8; ++i) {
        int kk = k0 + ((i < 4) ? 2 * i : 16 + 2 * (i - 4)) + 8 * hf;
        b.u[i] = *(const unsigned int*)(W + (size_t)(tile_n + t * 16 + l16) * K + kk);
      }
      acc[t] = __builtin_amdgcn_wmma_f32_16x16x32_bf16(
          false, a.v, false, b.v, (short)0, acc[t], false, false);
    }
  }

#pragma unroll
  for (int t = 0; t < 4; ++t) {
    int n = tile_n + t * 16 + l16;
#pragma unroll
    for (int r = 0; r < 8; ++r) {
      int m = tile_m + r + 8 * hf;
      float val = acc[t][r];
      if (mode == 0) {
        int h = n >> 6, d = n & 63;
        outb[((size_t)h * M + m) * HD + d] = f2bf(val);
      } else if (mode == 1) {
        int h = n >> 6, d = n & 63;
        outb[((size_t)h * HD + d) * M + m] = f2bf(val);
      } else {
        outf[(size_t)m * Nn + n] = val;
      }
    }
  }
}

// --------------------------- flash attention (per head) --------------------
// Q,K: [H][L][64] bf16, Vt: [H][64][L] bf16, Ob: [L][1024] bf16.
// Block = 4 waves (128 thr); wave owns 16 query rows; grid = (L/64, H).
// Per 32-key block: 4 score WMMAs + online softmax + 4 PV WMMAs.
__global__ __launch_bounds__(128) void flash_attn_k(
    const unsigned short* __restrict__ Qb, const unsigned short* __restrict__ Kb,
    const unsigned short* __restrict__ Vtb, unsigned short* __restrict__ Ob) {
  __shared__ unsigned short pbuf[4][16 * 32];   // per-wave P staging (C->A layout)

  const int lane = threadIdx.x & 31;
  const int wave = threadIdx.x >> 5;
  const int hf   = lane >> 4;
  const int l16  = lane & 15;
  const int h    = blockIdx.y;
  const int qw   = blockIdx.x * 64 + wave * 16;   // this wave's query-row base

  const unsigned short* Q  = Qb  + (size_t)h * SEQ_L * HD;
  const unsigned short* Kh = Kb  + (size_t)h * SEQ_L * HD;
  const unsigned short* Vt = Vtb + (size_t)h * HD * SEQ_L;

  // Q A-fragments for the two 32-wide slices of hd=64
  AFrag qa[2];
#pragma unroll
  for (int s = 0; s < 2; ++s)
#pragma unroll
    for (int i = 0; i < 8; ++i) {
      int d = 32 * s + ((i < 4) ? 2 * i : 16 + 2 * (i - 4)) + 8 * hf;
      qa[s].u[i] = *(const unsigned int*)(Q + (size_t)(qw + l16) * HD + d);
    }

  const v8f vzero = {0.f,0.f,0.f,0.f,0.f,0.f,0.f,0.f};
  v8f o[4]; o[0] = vzero; o[1] = vzero; o[2] = vzero; o[3] = vzero;
  float mrow[8], lrow[8];
#pragma unroll
  for (int r = 0; r < 8; ++r) { mrow[r] = -3.0e38f; lrow[r] = 0.f; }

  for (int j = 0; j < qw + 16; j += 32) {        // causal limit (uniform per wave)
    // ---- scores: two 16x16 tiles (keys j..j+15 and j+16..j+31) ----
    v8f s2[2]; s2[0] = vzero; s2[1] = vzero;
#pragma unroll
    for (int t = 0; t < 2; ++t) {
#pragma unroll
      for (int step = 0; step < 2; ++step) {
        AFrag bk;   // B[d][n] = K[j + t*16 + n][d]
#pragma unroll
        for (int i = 0; i < 8; ++i) {
          int d = 32 * step + ((i < 4) ? 2 * i : 16 + 2 * (i - 4)) + 8 * hf;
          bk.u[i] = *(const unsigned int*)(Kh + (size_t)(j + t * 16 + l16) * HD + d);
        }
        s2[t] = __builtin_amdgcn_wmma_f32_16x16x32_bf16(
            false, qa[step].v, false, bk.v, (short)0, s2[t], false, false);
      }
    }
    // ---- scale + causal mask ----
#pragma unroll
    for (int t = 0; t < 2; ++t)
#pragma unroll
      for (int r = 0; r < 8; ++r) {
        float v = s2[t][r] * 0.125f;             // 1/sqrt(64)
        int key = j + t * 16 + l16;
        int qr  = qw + r + 8 * hf;
        s2[t][r] = (key <= qr) ? v : -3.0e38f;
      }
    // ---- online softmax stats (rows live across the 16 lanes of a half) ----
#pragma unroll
    for (int r = 0; r < 8; ++r) {
      float mx = fmaxf(s2[0][r], s2[1][r]);
#pragma unroll
      for (int off = 1; off < 16; off <<= 1) mx = fmaxf(mx, __shfl_xor(mx, off));
      float mnew  = fmaxf(mrow[r], mx);
      float alpha = __expf(mrow[r] - mnew);
      float p0 = __expf(s2[0][r] - mnew);
      float p1 = __expf(s2[1][r] - mnew);
      s2[0][r] = p0; s2[1][r] = p1;
      float ls = p0 + p1;
#pragma unroll
      for (int off = 1; off < 16; off <<= 1) ls += __shfl_xor(ls, off);
      lrow[r] = lrow[r] * alpha + ls;
      mrow[r] = mnew;
#pragma unroll
      for (int t2 = 0; t2 < 4; ++t2) o[t2][r] *= alpha;
    }
    // ---- P: C-layout -> A-layout via per-wave LDS tile (same-wave ordering) ----
#pragma unroll
    for (int t = 0; t < 2; ++t)
#pragma unroll
      for (int r = 0; r < 8; ++r)
        pbuf[wave][(r + 8 * hf) * 32 + t * 16 + l16] = f2bf(s2[t][r]);
    asm volatile("s_wait_dscnt 0" ::: "memory");
    AFrag pa;
#pragma unroll
    for (int i = 0; i < 8; ++i) {
      int kb = ((i < 4) ? 2 * i : 16 + 2 * (i - 4)) + 8 * hf;
      pa.u[i] = *(const unsigned int*)(&pbuf[wave][l16 * 32 + kb]);
    }
    // ---- O += P * V  (contraction over 32 keys; Vt gives contiguous K-pairs) --
#pragma unroll
    for (int t2 = 0; t2 < 4; ++t2) {
      AFrag bv;   // B[k][n] = Vt[t2*16 + n][j + k]
#pragma unroll
      for (int i = 0; i < 8; ++i) {
        int kk = ((i < 4) ? 2 * i : 16 + 2 * (i - 4)) + 8 * hf;
        bv.u[i] = *(const unsigned int*)(Vt + (size_t)(t2 * 16 + l16) * SEQ_L + j + kk);
      }
      o[t2] = __builtin_amdgcn_wmma_f32_16x16x32_bf16(
          false, pa.v, false, bv.v, (short)0, o[t2], false, false);
    }
  }

  // ---- epilogue: normalize, store bf16 [L][D] with head offset ----
#pragma unroll
  for (int t2 = 0; t2 < 4; ++t2) {
    int d = t2 * 16 + l16;
#pragma unroll
    for (int r = 0; r < 8; ++r) {
      int m = qw + r + 8 * hf;
      float val = o[t2][r] / lrow[r];
      Ob[(size_t)m * DIM_D + h * HD + d] = f2bf(val);
    }
  }
}

// ---------------------------------------------------------------------------
extern "C" void kernel_launch(void* const* d_in, const int* in_sizes, int n_in,
                              void* d_out, int out_size, void* d_ws, size_t ws_size,
                              hipStream_t stream) {
  const float* x  = (const float*)d_in[0];
  const float* Wq = (const float*)d_in[1];
  const float* Wk = (const float*)d_in[2];
  const float* Wv = (const float*)d_in[3];
  const float* Wo = (const float*)d_in[4];
  float* out = (float*)d_out;

  const size_t LD = (size_t)SEQ_L * DIM_D;   // 4,194,304
  const size_t DD = (size_t)DIM_D * DIM_D;   // 1,048,576

  unsigned short* ws  = (unsigned short*)d_ws;   // 48 MB used
  unsigned short* xb  = ws;            // [L][D]
  unsigned short* wqb = xb  + LD;      // [D][D]
  unsigned short* wkb = wqb + DD;
  unsigned short* wvb = wkb + DD;
  unsigned short* wob = wvb + DD;
  unsigned short* Qb  = wob + DD;      // [H][L][64]
  unsigned short* Kb  = Qb  + LD;      // [H][L][64]
  unsigned short* Vtb = Kb  + LD;      // [H][64][L]
  unsigned short* Ob  = Vtb + LD;      // [L][D]

  // 1) convert fp32 -> bf16
  cvt_f32_to_bf16_k<<<(int)((LD + 255) / 256), 256, 0, stream>>>(x,  xb,  (int)LD);
  cvt_f32_to_bf16_k<<<(int)((DD + 255) / 256), 256, 0, stream>>>(Wq, wqb, (int)DD);
  cvt_f32_to_bf16_k<<<(int)((DD + 255) / 256), 256, 0, stream>>>(Wk, wkb, (int)DD);
  cvt_f32_to_bf16_k<<<(int)((DD + 255) / 256), 256, 0, stream>>>(Wv, wvb, (int)DD);
  cvt_f32_to_bf16_k<<<(int)((DD + 255) / 256), 256, 0, stream>>>(Wo, wob, (int)DD);

  // 2) QKV projections (WMMA, pipelined TDM-staged A tiles)
  dim3 ggrid(DIM_D / 256, SEQ_L / 16);
  gemm_bf16_k<<<ggrid, 128, 8192, stream>>>(xb, wqb, SEQ_L, DIM_D, DIM_D, 0, Qb,  nullptr);
  gemm_bf16_k<<<ggrid, 128, 8192, stream>>>(xb, wkb, SEQ_L, DIM_D, DIM_D, 0, Kb,  nullptr);
  gemm_bf16_k<<<ggrid, 128, 8192, stream>>>(xb, wvb, SEQ_L, DIM_D, DIM_D, 1, Vtb, nullptr);

  // 3) causal flash attention (WMMA)
  flash_attn_k<<<dim3(SEQ_L / 64, N_H), 128, 0, stream>>>(Qb, Kb, Vtb, Ob);

  // 4) output projection, fp32 store (WMMA)
  gemm_bf16_k<<<ggrid, 128, 8192, stream>>>(Ob, wob, SEQ_L, DIM_D, DIM_D, 2, nullptr, out);
}
